// MatchLSTMAttention_67551245632017
// MI455X (gfx1250) — compile-verified, matrix operand order checked by reading, not run on hardware
//
#include <hip/hip_runtime.h>

// ---------- types for WMMA ----------
typedef __bf16 bf16;
typedef bf16  v16bf __attribute__((ext_vector_type(16)));
typedef float v8f   __attribute__((ext_vector_type(8)));

// ---------- helpers ----------
static __device__ __forceinline__ unsigned short f32_to_bf16_rne(float f) {
    unsigned int u = __float_as_uint(f);
    unsigned int r = u + 0x7fffu + ((u >> 16) & 1u);   // round-to-nearest-even
    return (unsigned short)(r >> 16);
}
static __device__ __forceinline__ unsigned int pack_bf16x2(float lo, float hi) {
    return (unsigned int)f32_to_bf16_rne(lo) | ((unsigned int)f32_to_bf16_rne(hi) << 16);
}
// Branch-free tanh: v_exp_f32 + v_rcp_f32, no EXEC-mask slow path.
// x->+inf: e=inf, rcp->0, result 1.  x->-inf: e->0, result -1.  NaN propagates.
static __device__ __forceinline__ float fast_tanh(float x) {
    float e = __expf(2.0f * x);
    return 1.0f - 2.0f * __builtin_amdgcn_rcpf(e + 1.0f);
}

// =====================================================================
// Kernel 0: pre-swizzle Wq (f32 [512,512], row o, col k) into bf16 WMMA
// B-operand fragments.  Layout: [nt(32)][kstep(16)][lane(32)][d(8)] dwords,
// each dword = bf16 pair (K, K+1) for:
//   o = nt*16 + (lane&15)
//   K = kstep*32 + (lane>>4)*8 + (d&3)*2 + (d>>2)*16
// (16-bit A/B register layout, CDNA5 ISA §7.12.2)
// =====================================================================
__global__ void pack_wq_kernel(const float* __restrict__ Wq,
                               unsigned int* __restrict__ wqp) {
    int id = blockIdx.x * 256 + threadIdx.x;           // 131072 total dwords
    int d     = id & 7;
    int lane  = (id >> 3) & 31;
    int kstep = (id >> 8) & 15;
    int nt    = id >> 12;
    int o = nt * 16 + (lane & 15);
    int K = (kstep << 5) + ((lane >> 4) << 3) + ((d & 3) << 1) + ((d >> 2) << 4);
    const float* src = Wq + (size_t)o * 512 + K;
    wqp[id] = pack_bf16x2(src[0], src[1]);
}

// =====================================================================
// Kernel 1: base[b,o] = input_p@Wp.T + h@Wr.T + bp + bq + br   (64x512)
// =====================================================================
__global__ __launch_bounds__(512) void base_kernel(
    const float* __restrict__ input_p, const float* __restrict__ h_tm1,
    const float* __restrict__ Wp, const float* __restrict__ Wr,
    const float* __restrict__ bp, const float* __restrict__ bq,
    const float* __restrict__ br, float* __restrict__ base) {
    int b = blockIdx.x, o = threadIdx.x;
    __shared__ float pr[512], hr[512];
    pr[o] = input_p[(size_t)b * 512 + o];
    hr[o] = h_tm1[(size_t)b * 512 + o];
    __syncthreads();
    float acc = bp[o] + bq[o] + br[o];
    const float* wpo = Wp + (size_t)o * 512;
    const float* wro = Wr + (size_t)o * 512;
    #pragma unroll 4
    for (int k = 0; k < 512; k += 4) {
        float4 wp4 = *(const float4*)(wpo + k);
        float4 wr4 = *(const float4*)(wro + k);
        acc += pr[k] * wp4.x + pr[k+1] * wp4.y + pr[k+2] * wp4.z + pr[k+3] * wp4.w;
        acc += hr[k] * wr4.x + hr[k+1] * wr4.y + hr[k+2] * wr4.z + hr[k+3] * wr4.w;
    }
    base[(size_t)b * 512 + o] = acc;
}

// =====================================================================
// Kernel 2 (main): fused  alpha_logit[b,t] = w . tanh(base[b] + q_row@Wq.T)
// One block = 64 t-rows (4 M-subtiles of 16), 8 waves; wave owns 4 N-tiles.
// Each B fragment (global, L2-resident) is reused across the 4 M-subtiles,
// cutting L2 traffic 4x vs a 16-row tile.  256 WMMAs per wave.
// =====================================================================
__global__ __launch_bounds__(256) void attn_scores_kernel(
    const float* __restrict__ input_q, const unsigned int* __restrict__ wqp,
    const float* __restrict__ base, const float* __restrict__ w,
    const float* __restrict__ match_b, float* __restrict__ alpha) {

    __shared__ __align__(32) unsigned int aPack[16384];  // 64 KB: [mt(4)][kstep(16)][lane(32)][8]
    __shared__ float wavePartial[8][64];

    int tid = threadIdx.x;
    int b   = blockIdx.x >> 5;            // T/64 = 32 tiles per b
    int t0  = (blockIdx.x & 31) << 6;

    // ---- stage A tile (64 t-rows x 512 K) into LDS, bf16 WMMA layout ----
    #pragma unroll 4
    for (int i = 0; i < 64; ++i) {
        int j     = tid + (i << 8);       // 0..16383
        int d     = j & 7;
        int lane0 = (j >> 3) & 31;
        int kstep = (j >> 8) & 15;
        int mt    = j >> 12;
        int M = (mt << 4) + (lane0 & 15);
        int K = (kstep << 5) + ((lane0 >> 4) << 3) + ((d & 3) << 1) + ((d >> 2) << 4);
        const float* src = input_q + (((size_t)b * 2048 + t0 + M) << 9) + K;
        aPack[j] = pack_bf16x2(src[0], src[1]);
    }
    __syncthreads();

    int wave = tid >> 5;
    int lane = tid & 31;
    const v16bf* aFragP = (const v16bf*)aPack;   // 2048 fragments, mt stride 512
    const v16bf* bFragP = (const v16bf*)wqp;

    float rowAcc[4][8];                   // lanes 0/16 carry row partials
    #pragma unroll
    for (int mt = 0; mt < 4; ++mt)
        #pragma unroll
        for (int r = 0; r < 8; ++r) rowAcc[mt][r] = 0.0f;

    for (int q = 0; q < 4; ++q) {
        int nt = wave * 4 + q;            // N-tile (o0 = nt*16)
        v8f acc[4];
        #pragma unroll
        for (int mt = 0; mt < 4; ++mt) acc[mt] = (v8f){};

        #pragma unroll 4
        for (int kstep = 0; kstep < 16; ++kstep) {
            v16bf bm = bFragP[(nt * 16 + kstep) * 32 + lane];   // loaded once
            #pragma unroll
            for (int mt = 0; mt < 4; ++mt) {
                v16bf a = aFragP[(mt << 9) + kstep * 32 + lane];
                acc[mt] = __builtin_amdgcn_wmma_f32_16x16x32_bf16(
                    false, a, false, bm, (short)0, acc[mt], false, false);
            }
        }
        // C layout: VGPR r -> M = r + (lane>>4)*8, N = lane&15
        int o = nt * 16 + (lane & 15);
        float bb = base[((size_t)b << 9) + o];
        float wv = w[o];
        #pragma unroll
        for (int mt = 0; mt < 4; ++mt) {
            #pragma unroll
            for (int r = 0; r < 8; ++r) {
                float g = fast_tanh(acc[mt][r] + bb);
                float p = g * wv;
                // reduce over N within each 16-lane half (wave32)
                p += __shfl_xor(p, 1);
                p += __shfl_xor(p, 2);
                p += __shfl_xor(p, 4);
                p += __shfl_xor(p, 8);
                rowAcc[mt][r] += p;       // meaningful in lanes 0 and 16
            }
        }
    }
    if ((lane & 15) == 0) {
        int half = lane >> 4;
        #pragma unroll
        for (int mt = 0; mt < 4; ++mt)
            #pragma unroll
            for (int r = 0; r < 8; ++r)
                wavePartial[wave][(mt << 4) + half * 8 + r] = rowAcc[mt][r];
    }
    __syncthreads();
    if (tid < 64) {                        // deterministic cross-wave combine
        float s = match_b[0];
        #pragma unroll
        for (int wv2 = 0; wv2 < 8; ++wv2) s += wavePartial[wv2][tid];
        alpha[(size_t)b * 2048 + t0 + tid] = s;
    }
}

// =====================================================================
// Kernel 3: softmax over T=2048, per batch row (in place on alpha buffer)
// =====================================================================
__global__ __launch_bounds__(256) void softmax_kernel(float* __restrict__ alpha) {
    int b = blockIdx.x, tid = threadIdx.x;
    __shared__ float red[256];
    float* row = alpha + (size_t)b * 2048;
    float v[8];
    float m = -3.402823466e38f;
    #pragma unroll
    for (int i = 0; i < 8; ++i) { v[i] = row[tid + (i << 8)]; m = fmaxf(m, v[i]); }
    red[tid] = m; __syncthreads();
    for (int s = 128; s > 0; s >>= 1) {
        if (tid < s) red[tid] = fmaxf(red[tid], red[tid + s]);
        __syncthreads();
    }
    m = red[0]; __syncthreads();
    float ssum = 0.0f;
    #pragma unroll
    for (int i = 0; i < 8; ++i) { v[i] = expf(v[i] - m); ssum += v[i]; }
    red[tid] = ssum; __syncthreads();
    for (int s = 128; s > 0; s >>= 1) {
        if (tid < s) red[tid] += red[tid + s];
        __syncthreads();
    }
    float inv = 1.0f / red[0];
    #pragma unroll
    for (int i = 0; i < 8; ++i) row[tid + (i << 8)] = v[i] * inv;
}

// =====================================================================
// Kernel 4: partial z over a 256-length T-chunk (deterministic partials)
// grid = 64*8 blocks, 256 threads, 2 q-columns per thread (float2 coalesced)
// =====================================================================
__global__ __launch_bounds__(256) void z_partial_kernel(
    const float* __restrict__ input_q, const float* __restrict__ alpha,
    float* __restrict__ zp) {
    int b = blockIdx.x >> 3, chunk = blockIdx.x & 7;
    int tid = threadIdx.x;
    int q0 = tid * 2;
    int tbase = chunk * 256;
    float ax = 0.0f, ay = 0.0f;
    for (int t = 0; t < 256; ++t) {
        float a = alpha[(size_t)b * 2048 + tbase + t];
        const float2 qv = *(const float2*)(input_q +
                           (((size_t)b * 2048 + tbase + t) << 9) + q0);
        ax += a * qv.x;
        ay += a * qv.y;
    }
    float* dst = zp + ((size_t)(b * 8 + chunk) << 9) + q0;
    dst[0] = ax; dst[1] = ay;
}

// =====================================================================
// Kernel 5: out[b] = concat(input_p[b], sum_c zPartial[b][c])
// =====================================================================
__global__ __launch_bounds__(512) void finalize_kernel(
    const float* __restrict__ input_p, const float* __restrict__ zp,
    float* __restrict__ out) {
    int b = blockIdx.x, tid = threadIdx.x;
    out[(size_t)b * 1024 + tid] = input_p[(size_t)b * 512 + tid];
    float s = 0.0f;
    #pragma unroll
    for (int c = 0; c < 8; ++c) s += zp[((size_t)(b * 8 + c) << 9) + tid];
    out[(size_t)b * 1024 + 512 + tid] = s;
}

// =====================================================================
extern "C" void kernel_launch(void* const* d_in, const int* in_sizes, int n_in,
                              void* d_out, int out_size, void* d_ws, size_t ws_size,
                              hipStream_t stream) {
    const float* input_p = (const float*)d_in[0];
    const float* input_q = (const float*)d_in[1];
    const float* h_tm1   = (const float*)d_in[2];
    const float* Wp      = (const float*)d_in[3];
    const float* bp      = (const float*)d_in[4];
    const float* Wq      = (const float*)d_in[5];
    const float* bq      = (const float*)d_in[6];
    const float* Wr      = (const float*)d_in[7];
    const float* br      = (const float*)d_in[8];
    const float* w       = (const float*)d_in[9];
    const float* match_b = (const float*)d_in[10];
    float* out = (float*)d_out;

    char* ws = (char*)d_ws;
    unsigned int* wqp = (unsigned int*)ws;                 // 512 KB packed Wq(bf16)
    float* base  = (float*)(ws + (512  << 10));            // 128 KB
    float* alpha = (float*)(ws + (640  << 10));            // 512 KB logits/softmax
    float* zp    = (float*)(ws + (1152 << 10));            // 1 MB partial z
    // total scratch used: ~2.2 MB

    pack_wq_kernel    <<<512,  256, 0, stream>>>(Wq, wqp);
    base_kernel       <<<64,   512, 0, stream>>>(input_p, h_tm1, Wp, Wr, bp, bq, br, base);
    attn_scores_kernel<<<2048, 256, 0, stream>>>(input_q, wqp, base, w, match_b, alpha);
    softmax_kernel    <<<64,   256, 0, stream>>>(alpha);
    z_partial_kernel  <<<512,  256, 0, stream>>>(input_q, alpha, zp);
    finalize_kernel   <<<64,   512, 0, stream>>>(input_p, zp, out);
}
